// EncoderLayer_37469294690862
// MI455X (gfx1250) — compile-verified, matrix operand order checked by reading, not run on hardware
//
#include <hip/hip_runtime.h>
#include <hip/hip_bf16.h>

// ---------------------------------------------------------------------------
// Encoder layer for MI455X (gfx1250, wave32, WMMA + async global->LDS DMA).
// ---------------------------------------------------------------------------

#define B_   8
#define S_   1024
#define D_   1024
#define H_   16
#define DK_  64
#define F_   4096
#define NEGV (-1.0e18f)

typedef __bf16 bf16;
typedef __attribute__((ext_vector_type(16))) __bf16 v16bf;
typedef __attribute__((ext_vector_type(8)))  float  v8f;

// CDNA5 async global->LDS copy, 16B per lane, tracked by ASYNCcnt.
// LDS operand = wave-relative LDS offset = low 32 bits of the flat pointer.
__device__ __forceinline__ void async_copy_b128(const bf16* gptr, bf16* lptr) {
  unsigned loff = (unsigned)(unsigned long long)lptr;
  unsigned long long gaddr = (unsigned long long)gptr;
  asm volatile("global_load_async_to_lds_b128 %0, %1, off"
               :: "v"(loff), "v"(gaddr) : "memory");
}
#define WAIT_ASYNC_0() asm volatile("s_wait_asynccnt 0x0" ::: "memory")
#define WAIT_ASYNC_8() asm volatile("s_wait_asynccnt 0x8" ::: "memory")

// Load one 16x32 bf16 A/B fragment for this lane from an LDS row.
// Element e (0..7)  <- K = kk + khalf*8 + e
// Element e (8..15) <- K = kk + 16 + khalf*8 + (e-8)
__device__ __forceinline__ v16bf frag_ld(const bf16* p, int khalf) {
  union { uint4 u[2]; v16bf v; } c;
  c.u[0] = *(const uint4*)(p + khalf * 8);
  c.u[1] = *(const uint4*)(p + 16 + khalf * 8);
  return c.v;
}

// ---------------------------------------------------------------------------
// Generic bf16 GEMM: C[M,N] = A[M,K] * Bt[N,K]^T  (+bias) (ReLU) (+res)
// A row r lives at A + r*a_rs + (r>>10)*a_bp   (im2col-over-padded-buffer)
// bf16 output row r at o_base + r*o_rs + (r>>10)*o_bp
// Block tile 128x128x64, 256 threads = 8 waves (2x4), wave tile 64x32.
// Double-buffered LDS, tiles filled by GLOBAL_LOAD_ASYNC_TO_LDS_B128.
// ---------------------------------------------------------------------------
__global__ __launch_bounds__(256) void gemm_bf16_kernel(
    const bf16* __restrict__ A, long a_rs, long a_bp,
    const bf16* __restrict__ Bt, int Kd, int Nd,
    const float* __restrict__ bias, const float* __restrict__ res, int relu,
    float* __restrict__ outF, bf16* __restrict__ outB,
    long o_rs, long o_bp, long o_base)
{
  extern __shared__ char smem[];
  bf16* base = (bf16*)smem;   // 2 x { As[128][72], Bs[128][72] }

  const int tid   = threadIdx.x;
  const int wv    = tid >> 5;
  const int lane  = tid & 31;
  const int khalf = lane >> 4;
  const int lrow  = lane & 15;
  const int waveM = wv & 1;     // 2 waves over M (64 rows each)
  const int waveN = wv >> 1;    // 4 waves over N (32 cols each)

  const long blockM = (long)blockIdx.y * 128;
  const long blockN = (long)blockIdx.x * 128;

  // Per-thread chunk coordinates for tile fills (16B chunks, coalesced).
  const int rlA = tid >> 3;           // 32 rows per pass, 4 passes => 128 rows
  const int ccA = (tid & 7) * 8;      // 8 chunks of 8 bf16 per 64-wide row

  // Issue one K-tile's async copies (8 instructions per wave).
  auto issue_tile = [&](int kt, bf16* Asb, bf16* Bsb) {
    #pragma unroll
    for (int t = 0; t < 4; ++t) {
      int rl = rlA + t * 32;
      long rg = blockM + rl;
      async_copy_b128(A + rg * a_rs + (rg >> 10) * a_bp + (long)kt * 64 + ccA,
                      Asb + rl * 72 + ccA);
      long rb = blockN + rl;
      async_copy_b128(Bt + rb * (long)Kd + (long)kt * 64 + ccA,
                      Bsb + rl * 72 + ccA);
    }
  };

  v8f acc[4][2];
  #pragma unroll
  for (int i = 0; i < 4; ++i)
    #pragma unroll
    for (int j = 0; j < 2; ++j)
      #pragma unroll
      for (int r = 0; r < 8; ++r) acc[i][j][r] = 0.0f;

  const int ktiles = Kd >> 6;
  issue_tile(0, base, base + 9216);

  for (int kt = 0; kt < ktiles; ++kt) {
    bf16* Asb = base + (kt & 1) * 18432;
    bf16* Bsb = Asb + 9216;
    if (kt + 1 < ktiles) {
      bf16* Asn = base + ((kt + 1) & 1) * 18432;
      issue_tile(kt + 1, Asn, Asn + 9216);
      WAIT_ASYNC_8();                 // oldest 8 (current tile) retired
    } else {
      WAIT_ASYNC_0();
    }
    __syncthreads();

    #pragma unroll
    for (int kk = 0; kk < 64; kk += 32) {
      v16bf af[4], bfr[2];
      #pragma unroll
      for (int mi = 0; mi < 4; ++mi)
        af[mi] = frag_ld(Asb + (waveM * 64 + mi * 16 + lrow) * 72 + kk, khalf);
      #pragma unroll
      for (int ni = 0; ni < 2; ++ni)
        bfr[ni] = frag_ld(Bsb + (waveN * 32 + ni * 16 + lrow) * 72 + kk, khalf);
      #pragma unroll
      for (int mi = 0; mi < 4; ++mi)
        #pragma unroll
        for (int ni = 0; ni < 2; ++ni)
          acc[mi][ni] = __builtin_amdgcn_wmma_f32_16x16x32_bf16(
              false, af[mi], false, bfr[ni], (short)0, acc[mi][ni], false, false);
    }
    __syncthreads();    // all waves done reading before buffer is refilled
  }

  // Epilogue: bias -> ReLU -> residual -> store (f32 and/or bf16)
  #pragma unroll
  for (int mi = 0; mi < 4; ++mi)
    #pragma unroll
    for (int ni = 0; ni < 2; ++ni)
      #pragma unroll
      for (int r = 0; r < 8; ++r) {
        long row = blockM + waveM * 64 + mi * 16 + khalf * 8 + r;
        long col = blockN + waveN * 32 + ni * 16 + lrow;
        float v = acc[mi][ni][r];
        if (bias) v += bias[col];
        if (relu) v = v > 0.0f ? v : 0.0f;
        if (res)  v += res[row * (long)Nd + col];
        if (outF) outF[row * (long)Nd + col] = v;
        if (outB) outB[o_base + row * o_rs + (row >> 10) * o_bp + col] = (bf16)v;
      }
}

// ---------------------------------------------------------------------------
// Flash attention for one (b,h, 128-query block). 8 waves, each owns 16 rows.
// qkv: bf16 [B*S][3072] = q|k|v per row (q pre-scaled by DK^-0.5 via Wq).
// Q/K tiles arrive via async-to-LDS; V is transposed through registers.
// ---------------------------------------------------------------------------
__global__ __launch_bounds__(256) void attn_kernel(
    const bf16* __restrict__ qkv, const int* __restrict__ mask,
    bf16* __restrict__ ctx)
{
  extern __shared__ char smem[];
  bf16* Qs = (bf16*)smem;         // [128][72]
  bf16* Ks = Qs + 128 * 72;       // [128][72]
  bf16* Vt = Ks + 128 * 72;       // [64][136]   (V transposed: [dk][key])
  bf16* Ps = Vt + 64 * 136;       // [128][136]  (wave-private row bands)

  const int tid   = threadIdx.x;
  const int wv    = tid >> 5;
  const int lane  = tid & 31;
  const int khalf = lane >> 4;
  const int lrow  = lane & 15;
  const int qblk  = blockIdx.x;         // 0..7
  const int bh    = blockIdx.y;         // 0..127
  const int b     = bh >> 4, h = bh & 15;

  const int rlA = tid >> 3;
  const int ccA = (tid & 7) * 8;

  // Async-load the 128x64 Q block.
  const bf16* qb = qkv + (long)(b * S_) * 3072 + h * 64;
  #pragma unroll
  for (int t = 0; t < 4; ++t) {
    int rl = rlA + t * 32;
    int s = qblk * 128 + rl;
    async_copy_b128(qb + (long)s * 3072 + ccA, Qs + rl * 72 + ccA);
  }

  float m_run[8], l_run[8];
  #pragma unroll
  for (int r = 0; r < 8; ++r) { m_run[r] = -3.0e38f; l_run[r] = 0.0f; }
  v8f o[4];
  #pragma unroll
  for (int t = 0; t < 4; ++t)
    #pragma unroll
    for (int r = 0; r < 8; ++r) o[t][r] = 0.0f;

  const int qr0 = wv * 16;
  const int* mrow = mask + b * S_;

  for (int kb = 0; kb < S_ / 128; ++kb) {
    __syncthreads();    // previous iteration done reading Ks/Vt
    // K tile async ([key][dk] is already the Bt layout for Q.K^T);
    // V tile through registers with on-the-fly transpose into Vt.
    #pragma unroll
    for (int t = 0; t < 4; ++t) {
      int rl = rlA + t * 32;
      long rowg = (long)(b * S_ + kb * 128 + rl) * 3072 + h * 64 + ccA;
      async_copy_b128(qkv + rowg + 1024, Ks + rl * 72 + ccA);
      uint4 vv = *(const uint4*)(qkv + rowg + 2048);
      const bf16* ve = (const bf16*)&vv;
      #pragma unroll
      for (int j = 0; j < 8; ++j) Vt[(ccA + j) * 136 + rl] = ve[j];
    }
    WAIT_ASYNC_0();     // also covers the initial Q copies on kb==0
    __syncthreads();

    // S = Q K^T  (8 key tiles x 2 k-steps of WMMA)
    v8f st[8];
    #pragma unroll
    for (int nt = 0; nt < 8; ++nt) {
      v8f a;
      #pragma unroll
      for (int r = 0; r < 8; ++r) a[r] = 0.0f;
      #pragma unroll
      for (int kk = 0; kk < 64; kk += 32) {
        v16bf qa = frag_ld(Qs + (qr0 + lrow) * 72 + kk, khalf);
        v16bf kf = frag_ld(Ks + (nt * 16 + lrow) * 72 + kk, khalf);
        a = __builtin_amdgcn_wmma_f32_16x16x32_bf16(false, qa, false, kf,
                                                    (short)0, a, false, false);
      }
      st[nt] = a;
    }

    // Key mask + per-row max (rows live in 16-lane halves -> xor 1,2,4,8).
    float tmax[8];
    #pragma unroll
    for (int r = 0; r < 8; ++r) tmax[r] = -3.0e38f;
    #pragma unroll
    for (int nt = 0; nt < 8; ++nt) {
      int mv = mrow[kb * 128 + nt * 16 + lrow];
      #pragma unroll
      for (int r = 0; r < 8; ++r) {
        float sv = mv ? NEGV : st[nt][r];
        st[nt][r] = sv;
        tmax[r] = fmaxf(tmax[r], sv);
      }
    }
    #pragma unroll
    for (int r = 0; r < 8; ++r) {
      float v = tmax[r];
      v = fmaxf(v, __shfl_xor(v, 1));
      v = fmaxf(v, __shfl_xor(v, 2));
      v = fmaxf(v, __shfl_xor(v, 4));
      v = fmaxf(v, __shfl_xor(v, 8));
      tmax[r] = v;
    }

    // Online-softmax rescale of running state.
    float psum[8];
    #pragma unroll
    for (int r = 0; r < 8; ++r) {
      float mn = fmaxf(m_run[r], tmax[r]);
      float sc = __expf(m_run[r] - mn);
      m_run[r] = mn;
      l_run[r] *= sc;
      #pragma unroll
      for (int t = 0; t < 4; ++t) o[t][r] *= sc;
      psum[r] = 0.0f;
    }

    // P = exp(S - m), staged via wave-private LDS band (C-layout -> A-layout).
    #pragma unroll
    for (int nt = 0; nt < 8; ++nt)
      #pragma unroll
      for (int r = 0; r < 8; ++r) {
        float p = __expf(st[nt][r] - m_run[r]);
        psum[r] += p;
        Ps[(qr0 + khalf * 8 + r) * 136 + nt * 16 + lrow] = (bf16)p;
      }
    #pragma unroll
    for (int r = 0; r < 8; ++r) {
      float v = psum[r];
      v += __shfl_xor(v, 1);
      v += __shfl_xor(v, 2);
      v += __shfl_xor(v, 4);
      v += __shfl_xor(v, 8);
      l_run[r] += v;
    }

    // O += P V  (4 dk tiles x 4 k-steps; LDS reads are wave-local, no barrier)
    #pragma unroll
    for (int dt = 0; dt < 4; ++dt)
      #pragma unroll
      for (int kk = 0; kk < 128; kk += 32) {
        v16bf pa = frag_ld(Ps + (qr0 + lrow) * 136 + kk, khalf);
        v16bf vb = frag_ld(Vt + (dt * 16 + lrow) * 136 + kk, khalf);
        o[dt] = __builtin_amdgcn_wmma_f32_16x16x32_bf16(false, pa, false, vb,
                                                        (short)0, o[dt], false, false);
      }
  }

  // Normalize and scatter ctx in [B,S,D] bf16 layout (head-interleaved).
  #pragma unroll
  for (int r = 0; r < 8; ++r) {
    float inv = 1.0f / l_run[r];
    int s = qblk * 128 + qr0 + khalf * 8 + r;
    long rowoff = (long)(b * S_ + s) * 1024 + h * 64;
    #pragma unroll
    for (int dt = 0; dt < 4; ++dt)
      ctx[rowoff + dt * 16 + lrow] = (bf16)(o[dt][r] * inv);
  }
}

// ---------------------------------------------------------------------------
// LayerNorm (one 1024-elem row per 256-thread block) -> bf16, with optional
// padded-output mapping: out_base + row*1024 + (row>>10)*o_bp.
// ---------------------------------------------------------------------------
__global__ __launch_bounds__(256) void ln_kernel(
    const float* __restrict__ x, const float* __restrict__ g,
    const float* __restrict__ be, bf16* __restrict__ out,
    long o_bp, long o_base)
{
  __shared__ float ssum[8], ssq[8];
  long row = blockIdx.x;
  const float* xr = x + row * D_;
  const int tid = threadIdx.x;
  float4 v = ((const float4*)xr)[tid];
  float s = v.x + v.y + v.z + v.w;
  float q = v.x * v.x + v.y * v.y + v.z * v.z + v.w * v.w;
  #pragma unroll
  for (int off = 1; off < 32; off <<= 1) {
    s += __shfl_xor(s, off);
    q += __shfl_xor(q, off);
  }
  if ((tid & 31) == 0) { ssum[tid >> 5] = s; ssq[tid >> 5] = q; }
  __syncthreads();
  float ts = 0.0f, tq = 0.0f;
  #pragma unroll
  for (int i = 0; i < 8; ++i) { ts += ssum[i]; tq += ssq[i]; }
  float mu   = ts * (1.0f / D_);
  float var  = tq * (1.0f / D_) - mu * mu;
  float rstd = rsqrtf(var + 1e-6f);
  bf16* orow = out + o_base + row * D_ + (row >> 10) * o_bp;
  int c0 = tid * 4;
  const float* ve = (const float*)&v;
  #pragma unroll
  for (int j = 0; j < 4; ++j) {
    int c = c0 + j;
    orow[c] = (bf16)((ve[j] - mu) * rstd * g[c] + be[c]);
  }
}

// f32 [R][C] -> bf16 [C][R] (weights pre-transposed to N-major for the GEMM),
// with optional scale (folds DK^-0.5 into Wq). 64x64 LDS tiles, coalesced.
__global__ __launch_bounds__(256) void transpose_f32_to_bf16(
    const float* __restrict__ in, bf16* __restrict__ out,
    int R, int C, float scale)
{
  __shared__ float tile[64][65];
  int bx = blockIdx.x, by = blockIdx.y, tid = threadIdx.x;
  #pragma unroll
  for (int i = 0; i < 16; ++i) {
    int e = tid + i * 256;
    int r = e >> 6, c = e & 63;
    tile[r][c] = in[(long)(by * 64 + r) * C + bx * 64 + c];
  }
  __syncthreads();
  #pragma unroll
  for (int i = 0; i < 16; ++i) {
    int e = tid + i * 256;
    int wr = e >> 6, wc = e & 63;
    out[(long)(bx * 64 + wr) * R + by * 64 + wc] = (bf16)(tile[wc][wr] * scale);
  }
}

// Zero the 2 conv-pad rows per batch of a padded [B][S+2][W] bf16 buffer.
__global__ void zero_pad_rows(bf16* buf, int W)
{
  int bb = blockIdx.x;                  // 0..2*B-1
  int batch = bb >> 1;
  long row = (long)batch * (S_ + 2) + ((bb & 1) ? (S_ + 1) : 0);
  bf16* p = buf + row * W;
  for (int i = threadIdx.x; i < W; i += blockDim.x) p[i] = (bf16)0.0f;
}

// ---------------------------------------------------------------------------
extern "C" void kernel_launch(void* const* d_in, const int* in_sizes, int n_in,
                              void* d_out, int out_size, void* d_ws, size_t ws_size,
                              hipStream_t stream)
{
  (void)in_sizes; (void)n_in; (void)out_size; (void)ws_size;
  const float* x     = (const float*)d_in[0];
  const int*   mask  = (const int*)d_in[1];
  const float* Wq    = (const float*)d_in[2];
  const float* Wk    = (const float*)d_in[3];
  const float* Wv    = (const float*)d_in[4];
  const float* Wo    = (const float*)d_in[5];
  const float* ln1_g = (const float*)d_in[6];
  const float* ln1_b = (const float*)d_in[7];
  const float* ln2_g = (const float*)d_in[8];
  const float* ln2_b = (const float*)d_in[9];
  const float* c1w   = (const float*)d_in[10];
  const float* c1b   = (const float*)d_in[11];
  const float* c2w   = (const float*)d_in[12];
  const float* c2b   = (const float*)d_in[13];

  char* ws = (char*)d_ws;
  size_t off = 0;
  auto alloc = [&](size_t bytes) -> char* {
    char* p = ws + off;
    off += (bytes + 255) & ~(size_t)255;
    return p;
  };

  bf16*  qkvT = (bf16*)alloc((size_t)3072 * 1024 * 2);   // [3072][1024] WqT|WkT|WvT
  bf16*  WoT  = (bf16*)alloc((size_t)1024 * 1024 * 2);   // [1024][1024]
  bf16*  C1T  = (bf16*)alloc((size_t)4096 * 3072 * 2);   // [4096][3072]
  bf16*  C2T  = (bf16*)alloc((size_t)1024 * 12288 * 2);  // [1024][12288]
  bf16*  xnb  = (bf16*)alloc((size_t)8192 * 1024 * 2);   // LN1 out (reused as ctx)
  bf16*  qkv  = (bf16*)alloc((size_t)8192 * 3072 * 2);   // fused QKV
  float* x1   = (float*)alloc((size_t)8192 * 1024 * 4);  // attn residual out
  bf16*  xn2p = (bf16*)alloc((size_t)8 * 1026 * 1024 * 2); // padded LN2 out
  bf16*  hp   = (bf16*)alloc((size_t)8 * 1026 * 4096 * 2); // padded conv1 out
  bf16*  ctxb = xnb;  // xnb is dead after the QKV GEMM

  const size_t gemm_lds = (size_t)2 * 2 * 128 * 72 * 2;  // double-buffered A+B
  const size_t attn_lds = (size_t)(128 * 72 + 128 * 72 + 64 * 136 + 128 * 136) * 2;

  // Weight conversion / transpose (DK^-0.5 folded into Wq).
  transpose_f32_to_bf16<<<dim3(16, 16), 256, 0, stream>>>(Wq, qkvT,                 1024, 1024, 0.125f);
  transpose_f32_to_bf16<<<dim3(16, 16), 256, 0, stream>>>(Wk, qkvT + 1024 * 1024,   1024, 1024, 1.0f);
  transpose_f32_to_bf16<<<dim3(16, 16), 256, 0, stream>>>(Wv, qkvT + 2 * 1024 * 1024, 1024, 1024, 1.0f);
  transpose_f32_to_bf16<<<dim3(16, 16), 256, 0, stream>>>(Wo, WoT, 1024, 1024, 1.0f);
  transpose_f32_to_bf16<<<dim3(64, 48), 256, 0, stream>>>(c1w, C1T, 3072, 4096, 1.0f);
  transpose_f32_to_bf16<<<dim3(16, 192), 256, 0, stream>>>(c2w, C2T, 12288, 1024, 1.0f);

  // LN1 -> bf16
  ln_kernel<<<8192, 256, 0, stream>>>(x, ln1_g, ln1_b, xnb, 0, 0);

  // Fused QKV projection: [8192,1024] x [1024,3072] -> qkv bf16
  gemm_bf16_kernel<<<dim3(24, 64), 256, gemm_lds, stream>>>(
      xnb, 1024, 0, qkvT, 1024, 3072,
      nullptr, nullptr, 0, nullptr, qkv, 3072, 0, 0);

  // Flash attention -> ctx bf16 [B,S,D]
  attn_kernel<<<dim3(8, 128), 256, attn_lds, stream>>>(qkv, mask, ctxb);

  // Output projection + residual: x1 = x + ctx @ Wo
  gemm_bf16_kernel<<<dim3(8, 64), 256, gemm_lds, stream>>>(
      ctxb, 1024, 0, WoT, 1024, 1024,
      nullptr, x, 0, x1, nullptr, 0, 0, 0);

  // LN2 into zero-padded conv input
  zero_pad_rows<<<16, 256, 0, stream>>>(xn2p, 1024);
  ln_kernel<<<8192, 256, 0, stream>>>(x1, ln2_g, ln2_b, xn2p, 2048, 1024);

  // conv1 as im2col GEMM: [8192,3072] x [3072,4096], +b, ReLU -> padded hp
  zero_pad_rows<<<16, 256, 0, stream>>>(hp, 4096);
  gemm_bf16_kernel<<<dim3(32, 64), 256, gemm_lds, stream>>>(
      xn2p, 1024, 2048, C1T, 3072, 4096,
      c1b, nullptr, 1, nullptr, hp, 4096, 8192, 4096);

  // conv2 as im2col GEMM: [8192,12288] x [12288,1024], +b, ReLU, +x1 -> out
  gemm_bf16_kernel<<<dim3(8, 64), 256, gemm_lds, stream>>>(
      hp, 4096, 8192, C2T, 12288, 1024,
      c2b, x1, 1, (float*)d_out, nullptr, 0, 0, 0);
}
